// EncoderTransformerConv_80015240725030
// MI455X (gfx1250) — matrix operand order repaired and checked by reading, not compiled, not run on hardware
//
#include <hip/hip_runtime.h>
#include <hip/hip_bf16.h>

typedef __attribute__((ext_vector_type(16))) _Float16 v16h;
typedef __attribute__((ext_vector_type(8)))  _Float16 v8h;
typedef __attribute__((ext_vector_type(8)))  float    v8f;

#define NNODES 50000
#define NEDGES 800000

// ---------------------------------------------------------------------------
// Utility: fill a u32 region (zero f32 accumulators / amax / denom)
// ---------------------------------------------------------------------------
__global__ void fill_u32_kernel(unsigned* __restrict__ p, long long n, unsigned v) {
  long long i = (long long)blockIdx.x * blockDim.x + threadIdx.x;
  long long stride = (long long)gridDim.x * blockDim.x;
  for (; i < n; i += stride) p[i] = v;
}

// ---------------------------------------------------------------------------
// WMMA GEMM:  out[M,F] = A[M,K] * W[K,F] + bias[F]   (f32 in, f16 MACs, f32 acc)
// grid.x = M/16 row tiles, grid.y = F/64 col tiles, block = 128 threads (4 waves)
// Each wave owns one 16x16 tile. Whole K resident in LDS:
//   As  [16][K]  row-major, stride K+8 halfs (272B for K=128) -> bank-rotated rows
//   BsT [64][K]  column-of-W major (transposed), same padded stride
// Every fragment = two 16B-aligned ds_load_b128 chunks.
// ---------------------------------------------------------------------------
template<int K, int F>
__global__ __launch_bounds__(128)
void gemm_bias_wmma_kernel(const float* __restrict__ A,
                           const float* __restrict__ W,
                           const float* __restrict__ bias,
                           float* __restrict__ out) {
  constexpr int SA = K + 8;                         // padded LDS stride (halfs)
  __shared__ __attribute__((aligned(16))) _Float16 As[16 * SA];
  __shared__ __attribute__((aligned(16))) _Float16 BsT[64 * SA];

  const int tid  = threadIdx.x;
  const int lane = tid & 31;
  const int wv   = tid >> 5;                        // wave 0..3 -> 16-col sub-tile
  const long long rowBase = (long long)blockIdx.x * 16;
  const int colB = blockIdx.y * 64;

  // Stage A (16 x K): coalesced global reads, f32->f16
#pragma unroll
  for (int i = tid; i < 16 * K; i += 128) {
    const int r = i / K, c = i % K;                 // K is power of two -> shifts
    As[r * SA + c] = (_Float16)A[(rowBase + r) * K + c];
  }
  // Stage B (K x 64) transposed into BsT[c][k]: coalesced global reads
#pragma unroll
  for (int i = tid; i < K * 64; i += 128) {
    const int k = i >> 6, c = i & 63;
    BsT[c * SA + k] = (_Float16)W[(long long)k * F + colB + c];
  }
  __syncthreads();

  const int mr = lane & 15;          // A row within tile / D column
  const int hi = lane >> 4;          // lane half
  const int kb = hi * 8;             // A-fragment K interleave
  const int nb = wv * 16 + mr;       // B/D column within 64-col block

  v8f acc = {};
#pragma unroll
  for (int k0 = 0; k0 < K; k0 += 32) {
    // A 16x32 fragment: lane<16 holds K={0..7,16..23}, lane>=16 K={8..15,24..31}
    const v8h a0 = *(const v8h*)&As[mr * SA + k0 + kb];
    const v8h a1 = *(const v8h*)&As[mr * SA + k0 + 16 + kb];
    // B 32x16 fragment: lane<16 holds K={0..7,16..23}, lane>=16 K={8..15,24..31}, N=lane&15
    const v8h b0 = *(const v8h*)&BsT[nb * SA + k0 + kb];
    const v8h b1 = *(const v8h*)&BsT[nb * SA + k0 + 16 + kb];
    const v16h a = __builtin_shufflevector(a0, a1, 0,1,2,3,4,5,6,7,8,9,10,11,12,13,14,15);
    const v16h b = __builtin_shufflevector(b0, b1, 0,1,2,3,4,5,6,7,8,9,10,11,12,13,14,15);
    acc = __builtin_amdgcn_wmma_f32_16x16x32_f16(
        /*neg_a=*/false, a, /*neg_b=*/false, b,
        /*c_mod=*/(short)0, acc, /*reuse_a=*/false, /*reuse_b=*/false);
  }

  // D layout: VGPR r -> row (r + 8*lane_half), column = lane&15 within wave tile
  const float bvv = bias[colB + nb];
  const long long rb = rowBase + (hi << 3);
#pragma unroll
  for (int r = 0; r < 8; ++r) {
    out[(rb + r) * F + colB + nb] = acc[r] + bvv;
  }
}

// ---------------------------------------------------------------------------
// Edge stage: order-preserving float<->uint encoding for atomicMax segment-max
// ---------------------------------------------------------------------------
__device__ inline unsigned f2ord(float f) {
  unsigned u = __float_as_uint(f);
  return (u & 0x80000000u) ? ~u : (u | 0x80000000u);
}
__device__ inline float ord2f(unsigned e) {
  return __uint_as_float((e & 0x80000000u) ? (e ^ 0x80000000u) : ~e);
}

// alpha[e,h] = dot(q[dst[e],h,:], k[src[e],h,:]) / sqrt(64); atomicMax amax[dst,h]
__global__ __launch_bounds__(256)
void edge_alpha_kernel(const float* __restrict__ Q,
                       const float* __restrict__ Kf,
                       const int* __restrict__ src,
                       const int* __restrict__ dst,
                       float* __restrict__ alpha,
                       unsigned* __restrict__ amaxU) {
  long long t = (long long)blockIdx.x * blockDim.x + threadIdx.x;
  if (t >= (long long)NEDGES * 2) return;
  const int e = (int)(t >> 1), h = (int)(t & 1);
  const int s = src[e], d = dst[e];
  const float4* q4 = (const float4*)(Q  + (long long)d * 128 + h * 64);
  const float4* k4 = (const float4*)(Kf + (long long)s * 128 + h * 64);
  float acc = 0.f;
#pragma unroll
  for (int i = 0; i < 16; ++i) {
    float4 a = q4[i], b = k4[i];
    acc += a.x * b.x + a.y * b.y + a.z * b.z + a.w * b.w;
  }
  acc *= 0.125f;   // 1/sqrt(64)
  alpha[t] = acc;
  atomicMax(&amaxU[d * 2 + h], f2ord(acc));
}

// ex = exp(alpha - amax[dst]); alpha <- ex (in place); denom[dst,h] += ex
__global__ __launch_bounds__(256)
void edge_softmax_kernel(float* __restrict__ alpha,
                         const unsigned* __restrict__ amaxU,
                         float* __restrict__ denom,
                         const int* __restrict__ dst) {
  long long t = (long long)blockIdx.x * blockDim.x + threadIdx.x;
  if (t >= (long long)NEDGES * 2) return;
  const int e = (int)(t >> 1), h = (int)(t & 1);
  const int d = dst[e];
  const float m  = ord2f(amaxU[d * 2 + h]);
  const float ex = __expf(alpha[t] - m);
  alpha[t] = ex;
  atomicAdd(&denom[d * 2 + h], ex);
}

// acc[dst,h,:] += v[src,h,:] * (ex / (denom[dst,h] + 1e-16))
__global__ __launch_bounds__(256)
void edge_message_kernel(const float* __restrict__ V,
                         const float* __restrict__ ex,
                         const float* __restrict__ denom,
                         const int* __restrict__ src,
                         const int* __restrict__ dst,
                         float* __restrict__ acc) {
  long long t = (long long)blockIdx.x * blockDim.x + threadIdx.x;
  if (t >= (long long)NEDGES * 2) return;
  const int e = (int)(t >> 1), h = (int)(t & 1);
  const int s = src[e], d = dst[e];
  const float attn = ex[t] / (denom[d * 2 + h] + 1e-16f);
  const float4* v4 = (const float4*)(V + (long long)s * 128 + h * 64);
  float* ap = acc + (long long)d * 128 + h * 64;
#pragma unroll
  for (int i = 0; i < 16; ++i) {
    float4 v = v4[i];
    atomicAdd(ap + i * 4 + 0, v.x * attn);
    atomicAdd(ap + i * 4 + 1, v.y * attn);
    atomicAdd(ap + i * 4 + 2, v.z * attn);
    atomicAdd(ap + i * 4 + 3, v.w * attn);
  }
}

// out[n,c] = relu( 0.5*(acc[n,0,c] + acc[n,1,c]) + S[n,c] )
__global__ __launch_bounds__(256)
void finalize_kernel(const float* __restrict__ acc,
                     const float* __restrict__ S,
                     float* __restrict__ out) {
  long long t = (long long)blockIdx.x * blockDim.x + threadIdx.x;
  if (t >= (long long)NNODES * 64) return;
  const int n = (int)(t >> 6), c = (int)(t & 63);
  float v = 0.5f * (acc[(long long)n * 128 + c] + acc[(long long)n * 128 + 64 + c]) + S[t];
  out[t] = v > 0.f ? v : 0.f;
}

// ---------------------------------------------------------------------------
// Host-side per-layer driver (K compile-time: 128 for layer 1, 64 for 2/3)
// ---------------------------------------------------------------------------
template<int K>
static void run_layer(const float* xin,
                      const float* Wq, const float* bq,
                      const float* Wk, const float* bk,
                      const float* Wv, const float* bv,
                      const float* Ws, const float* bs,
                      const int* src, const int* dst,
                      float* Q, float* Kbuf, float* Vbuf, float* S,
                      float* alpha, unsigned* amaxU, float* denom,
                      float* out, hipStream_t stream) {
  const dim3 blk(128);
  const dim3 gQKV(NNODES / 16, 2);   // F=128 -> 2 col blocks
  const dim3 gS(NNODES / 16, 1);     // F=64  -> 1 col block
  gemm_bias_wmma_kernel<K, 128><<<gQKV, blk, 0, stream>>>(xin, Wq, bq, Q);
  gemm_bias_wmma_kernel<K, 128><<<gQKV, blk, 0, stream>>>(xin, Wk, bk, Kbuf);
  gemm_bias_wmma_kernel<K, 128><<<gQKV, blk, 0, stream>>>(xin, Wv, bv, Vbuf);
  gemm_bias_wmma_kernel<K, 64 ><<<gS,   blk, 0, stream>>>(xin, Ws, bs, S);

  // zero amax (ordered-encoding identity = 0) and denom
  fill_u32_kernel<<<512, 256, 0, stream>>>(amaxU, (long long)NNODES * 2, 0u);
  fill_u32_kernel<<<512, 256, 0, stream>>>((unsigned*)denom, (long long)NNODES * 2, 0u);

  const long long totEH = (long long)NEDGES * 2;
  const int egrid = (int)((totEH + 255) / 256);
  edge_alpha_kernel<<<egrid, 256, 0, stream>>>(Q, Kbuf, src, dst, alpha, amaxU);

  // acc aliases Q: Q is dead after edge_alpha; zero it for accumulation
  fill_u32_kernel<<<2048, 256, 0, stream>>>((unsigned*)Q, (long long)NNODES * 128, 0u);

  edge_softmax_kernel<<<egrid, 256, 0, stream>>>(alpha, amaxU, denom, dst);
  edge_message_kernel<<<egrid, 256, 0, stream>>>(Vbuf, alpha, denom, src, dst, Q);

  const int fgrid = (int)(((long long)NNODES * 64 + 255) / 256);
  finalize_kernel<<<fgrid, 256, 0, stream>>>(Q, S, out);
}

extern "C" void kernel_launch(void* const* d_in, const int* in_sizes, int n_in,
                              void* d_out, int out_size, void* d_ws, size_t ws_size,
                              hipStream_t stream) {
  (void)in_sizes; (void)n_in; (void)out_size; (void)ws_size;

  const float* x  = (const float*)d_in[0];
  const int*   ei = (const int*)d_in[1];
  const int* src = ei;            // edge_index[0, :]
  const int* dst = ei + NEDGES;   // edge_index[1, :]

  // inputs 2..25: per layer L: Wq,bq,Wk,bk,Wv,bv,Ws,bs
  const float* W[3][4];
  const float* B[3][4];
  int idx = 2;
  for (int L = 0; L < 3; ++L)
    for (int j = 0; j < 4; ++j) {
      W[L][j] = (const float*)d_in[idx++];
      B[L][j] = (const float*)d_in[idx++];
    }

  // Workspace carve-up (floats). acc aliases Q inside run_layer.
  float* ws = (float*)d_ws;
  size_t off = 0;
  float*    Q     = ws + off; off += (size_t)NNODES * 128;
  float*    Kbuf  = ws + off; off += (size_t)NNODES * 128;
  float*    Vbuf  = ws + off; off += (size_t)NNODES * 128;
  float*    S     = ws + off; off += (size_t)NNODES * 64;
  float*    alpha = ws + off; off += (size_t)NEDGES * 2;
  unsigned* amaxU = (unsigned*)(ws + off); off += (size_t)NNODES * 2;
  float*    denom = ws + off; off += (size_t)NNODES * 2;
  float*    hbuf  = ws + off; off += (size_t)NNODES * 64;

  // Layer 1: K=128, input x -> hbuf
  run_layer<128>(x,
                 W[0][0], B[0][0], W[0][1], B[0][1], W[0][2], B[0][2], W[0][3], B[0][3],
                 src, dst, Q, Kbuf, Vbuf, S, alpha, amaxU, denom, hbuf, stream);
  // Layer 2: K=64, hbuf -> hbuf (GEMMs consume hbuf before finalize rewrites it)
  run_layer<64>(hbuf,
                W[1][0], B[1][0], W[1][1], B[1][1], W[1][2], B[1][2], W[1][3], B[1][3],
                src, dst, Q, Kbuf, Vbuf, S, alpha, amaxU, denom, hbuf, stream);
  // Layer 3: K=64, hbuf -> d_out
  run_layer<64>(hbuf,
                W[2][0], B[2][0], W[2][1], B[2][1], W[2][2], B[2][2], W[2][3], B[2][3],
                src, dst, Q, Kbuf, Vbuf, S, alpha, amaxU, denom, (float*)d_out, stream);
}
